// HybridQLSTM_65481071408905
// MI455X (gfx1250) — compile-verified, matrix operand order checked by reading, not run on hardware
//
#include <hip/hip_runtime.h>
#include <hip/hip_bf16.h>

// ---- problem constants (match reference) ----
#define BB   8
#define SS   4096
#define EE   256
#define HH   256
#define G4   1024        // 4*H
#define TT   1000
#define BS   32768       // B*S

typedef __attribute__((ext_vector_type(16))) _Float16 v16h;
typedef __attribute__((ext_vector_type(8)))  float    v8f;
typedef __attribute__((ext_vector_type(4)))  int      i4;

// CDNA5 feature probes (device pass only; host pass sees 0 and fallback code)
#if defined(__HIP_DEVICE_COMPILE__)
  #if __has_builtin(__builtin_amdgcn_global_load_async_to_lds_b128)
    #define HAVE_ASYNC 1
  #endif
  #if __has_builtin(__builtin_amdgcn_tensor_load_to_lds)
    #define HAVE_TDM 1
  #endif
#endif
#ifndef HAVE_ASYNC
#define HAVE_ASYNC 0
#endif
#ifndef HAVE_TDM
#define HAVE_TDM 0
#endif

// address-space-qualified int4, matching the async builtin's prototype
typedef __attribute__((address_space(1))) i4 i4_g;   // global
typedef __attribute__((address_space(3))) i4 i4_l;   // LDS
typedef __attribute__((address_space(3))) void LV;   // LDS void (for TDM addr)

// WMMA wrapper: D = A(16x32 f16) * B(32x16 f16) + C(16x16 f32)
__device__ __forceinline__ v8f wmma_f16(v16h a, v16h b, v8f c) {
    return __builtin_amdgcn_wmma_f32_16x16x32_f16(
        /*neg_a=*/false, a, /*neg_b=*/false, b,
        /*c_mod=*/(short)0, c, /*reuse_a=*/false, /*reuse_b=*/false);
}

// Load a 16x32 f16 A-fragment from LDS (row-major, row stride = lda halfs).
// ISA layout: lanes 0-15 -> M=lane, Kbase=0; lanes 16-31 -> M=lane-16, Kbase=8.
// halfs 0..7 = K kb+0..7, halfs 8..15 = K kb+16..23.
__device__ __forceinline__ v16h load_a_frag(const _Float16* lds, int lda, int k0) {
    const int lane = threadIdx.x & 31;
    const int li   = lane & 15;
    const int kb   = (lane >> 4) << 3;
    const _Float16* q = lds + li * lda + k0 + kb;
    v16h a;
#pragma unroll
    for (int j = 0; j < 8; ++j) a[j] = q[j];
#pragma unroll
    for (int j = 0; j < 8; ++j) a[8 + j] = q[16 + j];
    return a;
}

// Load a 32x16 f16 B-fragment with on-the-fly f32->f16 conversion.
// B[k,n] = W[n*ldw + k] (i.e. B = W^T), column n = ncol for this lane.
__device__ __forceinline__ v16h load_b_frag_f32(const float* W, int ldw, int ncol, int k0) {
    const int lane = threadIdx.x & 31;
    const int kb   = (lane >> 4) << 3;
    const float* w = W + (size_t)ncol * ldw + k0 + kb;
    v16h b;
#pragma unroll
    for (int j = 0; j < 8; ++j) b[j] = (_Float16)w[j];
#pragma unroll
    for (int j = 0; j < 8; ++j) b[8 + j] = (_Float16)w[16 + j];
    return b;
}

// =====================================================================
// Kernel 1: embedding gather + input projection
//   xg[s, b, g] = emb[sent[b,s], :] @ W_ih[g, :]^T + b_ih[g] + b_hh[g]
// (stored time-major so the recurrent kernel streams one contiguous
//  32KB slice per step)
// grid = BS/16 blocks of 256 threads (8 waves); block tile = 16(M) x 1024(N)
// =====================================================================
__global__ __launch_bounds__(256) void lstm_inproj(
    const int* __restrict__ sent, const float* __restrict__ emb,
    const float* __restrict__ W_ih, const float* __restrict__ b_ih,
    const float* __restrict__ b_hh, float* __restrict__ xg)
{
    __shared__ _Float16 Alds[16 * EE];

    const int mbase = blockIdx.x * 16;
    const int tid   = threadIdx.x;

    // gather 16 embedding rows into LDS as f16
    for (int i = tid; i < 16 * EE; i += 256) {
        const int mm  = i >> 8;
        const int e   = i & 255;
        const int tok = sent[mbase + mm];
        Alds[i] = (_Float16)emb[(size_t)tok * EE + e];
    }
    __syncthreads();

    const int wave = tid >> 5;
    const int lane = tid & 31;
    const int li   = lane & 15;
    const int mrow = (lane >> 4) << 3;

    // each wave covers 8 N-tiles: n0 = wave*128 + t*16
    v8f acc[8];
#pragma unroll
    for (int t = 0; t < 8; ++t) {
        const int n = wave * 128 + t * 16 + li;
        const float bias = b_ih[n] + b_hh[n];
#pragma unroll
        for (int j = 0; j < 8; ++j) acc[t][j] = bias;
    }

    for (int kc = 0; kc < 8; ++kc) {
        const int k0 = kc * 32;
        const v16h a = load_a_frag(Alds, EE, k0);     // shared across 8 N-tiles
#pragma unroll
        for (int t = 0; t < 8; ++t) {
            const int n = wave * 128 + t * 16 + li;
            const v16h b = load_b_frag_f32(W_ih, EE, n, k0);
            acc[t] = wmma_f16(a, b, acc[t]);
        }
    }

#pragma unroll
    for (int t = 0; t < 8; ++t) {
        const int n = wave * 128 + t * 16 + li;
#pragma unroll
        for (int j = 0; j < 8; ++j) {
            const int m  = mbase + mrow + j;
            const int bb = m >> 12;        // m / S
            const int ss = m & (SS - 1);   // m % S
            xg[((size_t)ss * BB + bb) * G4 + n] = acc[t][j];
        }
    }
}

// =====================================================================
// Kernel 2: persistent recurrent LSTM. One workgroup (32 waves, 1024 thr).
// W_hh kept as f16 B-fragments in registers (16 frags / wave = its 32 cols).
// h (f16) + gate buffer (f32) in LDS; c in 2 registers/thread.
// Per step: prefetch xg slice s+4 (L2), async-DMA slice s into LDS,
// 16 WMMAs, elementwise gate update. 4096 sequential steps.
// =====================================================================
__global__ __launch_bounds__(1024) void lstm_recurrent(
    const float* __restrict__ W_hh, const float* __restrict__ xg,
    _Float16* __restrict__ hs)
{
    __shared__ _Float16 Hld[16 * HH];   // h, rows 8..15 stay zero (M padding)
    __shared__ float    Gld[BB * G4];   // xg slice, then gate pre-activations

    const int tid  = threadIdx.x;
    const int wave = tid >> 5;
    const int lane = tid & 31;
    const int li   = lane & 15;
    const int mrow = (lane >> 4) << 3;

    for (int i = tid; i < 16 * HH; i += 1024) Hld[i] = (_Float16)0.f;
    float creg[2] = {0.f, 0.f};         // cell state, owned per-thread

    // preload this wave's W_hh^T fragments (columns n = wave*32 .. wave*32+31)
    v16h Bf[8][2];
#pragma unroll
    for (int kc = 0; kc < 8; ++kc) {
#pragma unroll
        for (int t = 0; t < 2; ++t) {
            const int n = wave * 32 + t * 16 + li;
            Bf[kc][t] = load_b_frag_f32(W_hh, HH, n, kc * 32);
        }
    }
    __syncthreads();

    for (int s = 0; s < SS; ++s) {
        const float* xslice = xg + (size_t)s * (BB * G4);

        // prefetch the slice 4 steps ahead into cache (global_prefetch_b8)
        if (tid < 256 && s + 4 < SS)
            __builtin_prefetch(xg + (size_t)(s + 4) * (BB * G4) + tid * 32, 0, 3);

#if HAVE_ASYNC
        // async DMA: 32KB xg slice -> LDS (2 x b128 per thread)
#pragma unroll
        for (int r = 0; r < 2; ++r) {
            const int off = (tid + r * 1024) * 4;   // float index, 16B chunks
            __builtin_amdgcn_global_load_async_to_lds_b128(
                (i4_g*)(i4*)(void*)(xslice + off),
                (i4_l*)(i4*)(void*)(Gld + off), 0, 0);
        }
        asm volatile("s_wait_asynccnt 0" ::: "memory");
        __syncthreads();
#endif

        // init accumulators from xg (contains both biases)
        v8f acc[2];
#pragma unroll
        for (int t = 0; t < 2; ++t) {
            const int n = wave * 32 + t * 16 + li;
#pragma unroll
            for (int j = 0; j < 8; ++j) {
                const int m = mrow + j;   // batch row; 8..15 are padding
#if HAVE_ASYNC
                acc[t][j] = (m < BB) ? Gld[m * G4 + n] : 0.f;
#else
                acc[t][j] = (m < BB) ? xslice[(size_t)m * G4 + n] : 0.f;
#endif
            }
        }
        // g += h @ W_hh^T
#pragma unroll
        for (int kc = 0; kc < 8; ++kc) {
            const v16h a = load_a_frag(Hld, HH, kc * 32);
            acc[0] = wmma_f16(a, Bf[kc][0], acc[0]);
            acc[1] = wmma_f16(a, Bf[kc][1], acc[1]);
        }
        // scatter gates (each element read+written by its owning thread only)
#pragma unroll
        for (int t = 0; t < 2; ++t) {
            const int n = wave * 32 + t * 16 + li;
#pragma unroll
            for (int j = 0; j < 8; ++j) {
                const int m = mrow + j;
                if (m < BB) Gld[m * G4 + n] = acc[t][j];
            }
        }
        __syncthreads();

        // elementwise LSTM cell update: 8*256 = 2048 items over 1024 threads
#pragma unroll
        for (int r = 0; r < 2; ++r) {
            const int idx = tid + r * 1024;
            const int b   = idx >> 8;
            const int j   = idx & 255;
            const float gi = Gld[b * G4 + j];
            const float gf = Gld[b * G4 + HH + j];
            const float gg = Gld[b * G4 + 2 * HH + j];
            const float go = Gld[b * G4 + 3 * HH + j];
            const float i_ = 1.f / (1.f + __expf(-gi));
            const float f_ = 1.f / (1.f + __expf(-gf));
            const float g_ = tanhf(gg);
            const float o_ = 1.f / (1.f + __expf(-go));
            const float c  = f_ * creg[r] + i_ * g_;
            const float h  = o_ * tanhf(c);
            creg[r] = c;
            Hld[b * HH + j] = (_Float16)h;
            hs[((size_t)b * SS + s) * HH + j] = (_Float16)h;
        }
        __syncthreads();
    }
}

// =====================================================================
// Kernel 3: output projection + fused log_softmax.
// grid = BS/16 blocks of 1024 threads (32 waves). Block tile: 16(M) x 1024(N),
// N padded 1000->1024 (pad columns masked out of softmax + output).
// A-tile (16x256 f16, contiguous 8KB) is fetched by the Tensor Data Mover.
// =====================================================================
__global__ __launch_bounds__(1024) void lstm_outproj(
    const _Float16* __restrict__ hs, const float* __restrict__ W_out,
    const float* __restrict__ b_out, float* __restrict__ out)
{
    __shared__ _Float16 Alds[16 * HH];
    __shared__ float    Red[16 * 32];
    __shared__ float    RowMax[16];
    __shared__ float    RowSum[16];

    const int mbase = blockIdx.x * 16;
    const int tid   = threadIdx.x;
    const int wave  = tid >> 5;
    const int lane  = tid & 31;
    const int li    = lane & 15;
    const int mrow  = (lane >> 4) << 3;

#if HAVE_TDM
    if (tid < 32) {   // wave 0 issues one TDM descriptor for the 8KB A-tile
        typedef unsigned int tv4u __attribute__((ext_vector_type(4)));
        typedef int          tv4i __attribute__((ext_vector_type(4)));
        typedef int          tv8i __attribute__((ext_vector_type(8)));
        const unsigned lds_base =
            (unsigned)(unsigned long long)(LV*)(void*)&Alds[0];
        const unsigned long long ga =
            (unsigned long long)(const void*)(hs + (size_t)mbase * HH);
        // D# group0: count=1 | lds_addr | global_addr | type=2
        tv4u g0 = { 1u,
                    lds_base,
                    (unsigned)(ga & 0xFFFFFFFFull),
                    (unsigned)((ga >> 32) & 0x1FFFFFFull) | (2u << 30) };
        // D# group1: data_size=8B; tensor_dim0=tile_dim0=1024 units (8KB, 1D);
        // tensor_dim1=1; dim0_stride=1024
        tv8i g1 = { (int)(3u << 16),
                    (int)(1024u << 16),
                    (int)(1u << 16),
                    (int)(1024u << 16),
                    0, 1024, 0, 0 };
        tv4i gz4 = { 0, 0, 0, 0 };
#if defined(__clang_major__) && (__clang_major__ >= 23)
        tv8i gz8 = { 0, 0, 0, 0, 0, 0, 0, 0 };
        __builtin_amdgcn_tensor_load_to_lds(g0, g1, gz4, gz4, gz8, 0);
#else
        __builtin_amdgcn_tensor_load_to_lds(g0, g1, gz4, gz4, 0);
#endif
        __builtin_amdgcn_s_wait_tensorcnt((short)0);
    }
    __syncthreads();
#else
    for (int i = tid; i < 16 * HH; i += 1024)
        Alds[i] = hs[(size_t)mbase * HH + i];
    __syncthreads();
#endif

    int ncol[2];
    v8f acc[2];
#pragma unroll
    for (int t = 0; t < 2; ++t) {
        const int n = wave * 32 + t * 16 + li;
        ncol[t] = n;
        const float bias = (n < TT) ? b_out[n] : 0.f;
#pragma unroll
        for (int j = 0; j < 8; ++j) acc[t][j] = bias;
    }

    for (int kc = 0; kc < 8; ++kc) {
        const int k0 = kc * 32;
        const v16h a = load_a_frag(Alds, HH, k0);
#pragma unroll
        for (int t = 0; t < 2; ++t) {
            v16h b;
            if (ncol[t] < TT) {
                b = load_b_frag_f32(W_out, HH, ncol[t], k0);
            } else {
#pragma unroll
                for (int j = 0; j < 16; ++j) b[j] = (_Float16)0.f;
            }
            acc[t] = wmma_f16(a, b, acc[t]);
        }
    }

    // ---- row max over N (valid cols only) ----
#pragma unroll
    for (int j = 0; j < 8; ++j) {
        float v0 = (ncol[0] < TT) ? acc[0][j] : -__builtin_inff();
        float v1 = (ncol[1] < TT) ? acc[1][j] : -__builtin_inff();
        float v  = fmaxf(v0, v1);
#pragma unroll
        for (int d = 8; d >= 1; d >>= 1) v = fmaxf(v, __shfl_xor(v, d, 32));
        if (li == 0) Red[(mrow + j) * 32 + wave] = v;   // lanes 0 & 16 write
    }
    __syncthreads();
    if (tid < 16) {
        float v = -__builtin_inff();
        for (int w = 0; w < 32; ++w) v = fmaxf(v, Red[tid * 32 + w]);
        RowMax[tid] = v;
    }
    __syncthreads();
    float rmax[8];
#pragma unroll
    for (int j = 0; j < 8; ++j) rmax[j] = RowMax[mrow + j];

    // ---- row sum of exp ----
#pragma unroll
    for (int j = 0; j < 8; ++j) {
        float v = 0.f;
        if (ncol[0] < TT) v += __expf(acc[0][j] - rmax[j]);
        if (ncol[1] < TT) v += __expf(acc[1][j] - rmax[j]);
#pragma unroll
        for (int d = 8; d >= 1; d >>= 1) v += __shfl_xor(v, d, 32);
        if (li == 0) Red[(mrow + j) * 32 + wave] = v;
    }
    __syncthreads();
    if (tid < 16) {
        float v = 0.f;
        for (int w = 0; w < 32; ++w) v += Red[tid * 32 + w];
        RowSum[tid] = v;
    }
    __syncthreads();

    // ---- write log-softmax ----
#pragma unroll
    for (int j = 0; j < 8; ++j) {
        const float lse = rmax[j] + __logf(RowSum[mrow + j]);
        const size_t mo = (size_t)(mbase + mrow + j) * TT;
#pragma unroll
        for (int t = 0; t < 2; ++t) {
            if (ncol[t] < TT) out[mo + ncol[t]] = acc[t][j] - lse;
        }
    }
}

extern "C" void kernel_launch(void* const* d_in, const int* in_sizes, int n_in,
                              void* d_out, int out_size, void* d_ws, size_t ws_size,
                              hipStream_t stream) {
    const int*   sent  = (const int*)d_in[0];
    const float* emb   = (const float*)d_in[1];
    const float* W_ih  = (const float*)d_in[2];
    const float* W_hh  = (const float*)d_in[3];
    const float* b_ih  = (const float*)d_in[4];
    const float* b_hh  = (const float*)d_in[5];
    const float* W_out = (const float*)d_in[6];
    const float* b_out = (const float*)d_in[7];
    float* out = (float*)d_out;

    // workspace layout: xg (f32, 128 MB, time-major) | hs (f16, 16 MB)
    float*     xg = (float*)d_ws;
    _Float16*  hs = (_Float16*)((char*)d_ws + (size_t)BS * G4 * sizeof(float));

    lstm_inproj   <<<BS / 16,  256, 0, stream>>>(sent, emb, W_ih, b_ih, b_hh, xg);
    lstm_recurrent<<<1,       1024, 0, stream>>>(W_hh, xg, hs);
    lstm_outproj  <<<BS / 16, 1024, 0, stream>>>(hs, W_out, b_out, out);
}
